// CustomMHA_58239756533774
// MI455X (gfx1250) — matrix operand order repaired and checked by reading, not compiled, not run on hardware
//
#include <hip/hip_runtime.h>
#include <hip/hip_bf16.h>
#include <math.h>

typedef _Float16 half_t;
typedef __attribute__((ext_vector_type(16))) _Float16 v16h;
typedef __attribute__((ext_vector_type(8)))  _Float16 v8h;
typedef __attribute__((ext_vector_type(8)))  float    v8f;
typedef __attribute__((ext_vector_type(4)))  int      v4i;
typedef __attribute__((address_space(1))) v4i as1_v4i;
typedef __attribute__((address_space(3))) v4i as3_v4i;

#define BATCH 4
#define SEQ   2048
#define DMODEL 1024
#define NHEADS 16
#define DHEAD 64
#define TD    3072   // 3*DMODEL

#ifndef __has_builtin
#define __has_builtin(x) 0
#endif

#if __has_builtin(__builtin_amdgcn_global_load_async_to_lds_b128) && \
    __has_builtin(__builtin_amdgcn_s_wait_asynccnt)
#define USE_ASYNC_LDS 1
// Copy 16 bytes global -> LDS via async DMA (ASYNCcnt-tracked).
#define ASYNC_B128(gptr, lptr)                                                  \
    __builtin_amdgcn_global_load_async_to_lds_b128(                             \
        (as1_v4i*)(unsigned long long)(gptr),                                   \
        (as3_v4i*)(unsigned int)(unsigned long long)(lptr),                     \
        0, 0)
#define ASYNC_WAIT0() __builtin_amdgcn_s_wait_asynccnt(0)
#define ASYNC_WAIT2() __builtin_amdgcn_s_wait_asynccnt(2)
#else
#define USE_ASYNC_LDS 0
#define ASYNC_B128(gptr, lptr) \
    do { *(v8h*)(lptr) = *(const v8h*)(gptr); } while (0)
#define ASYNC_WAIT0() do {} while (0)
#define ASYNC_WAIT2() do {} while (0)
#endif

// ---------------------------------------------------------------------------
// fp32 -> f16 elementwise convert
// ---------------------------------------------------------------------------
__global__ void mha_cvt_f32_to_f16(const float* __restrict__ src,
                                   half_t* __restrict__ dst, int n) {
    int i = blockIdx.x * blockDim.x + threadIdx.x;
    if (i < n) dst[i] = (half_t)src[i];
}

// ---------------------------------------------------------------------------
// Repack row-major fp32 W[K x N] into WMMA B-fragment order, f16, grouped so
// that the 4 N-subtiles of a 64-wide N-group are contiguous per k-chunk:
//   dst[(((ng*(K/32) + kt)*4 + j)*512) + lane*16 + i]
// Lane L of tile (nt=ng*4+j, kt): col = nt*16 + (L&15),
// K rows kt*32 + 16*(L>>4) + i.
// grid = (K/32, N/16), block = 32
// ---------------------------------------------------------------------------
__global__ __launch_bounds__(32)
void mha_repack_b(const float* __restrict__ W, half_t* __restrict__ P,
                  int K, int N) {
    int kt = blockIdx.x, nt = blockIdx.y, lane = threadIdx.x;
    int ng = nt >> 2, j = nt & 3;
    int c  = lane & 15;
    int kb = kt * 32 + ((lane >> 4) << 4);
    v16h v;
#pragma unroll
    for (int i = 0; i < 16; ++i)
        v[i] = (half_t)W[(size_t)(kb + i) * N + nt * 16 + c];
    size_t base = (((size_t)ng * (K >> 5) + kt) * 4 + j) * 512;
    *(v16h*)(P + base + lane * 16) = v;
}

// ---------------------------------------------------------------------------
// GEMM: C[M x N] = A[M x K](f16, row-major) * Bpacked.
// Block = 4 waves = 64x64 output tile (wave w -> rows +w*16, all 64 cols).
// B chunk (4KB per k-step) staged to double-buffered LDS via async DMA and
// shared by all 4 waves; each A fragment feeds 4 WMMAs. All 4 B fragments
// are preloaded into registers so the WMMAs issue back-to-back after a
// single DScnt wait.
// grid = (M/64, N/64), block = 128.
// ---------------------------------------------------------------------------
__global__ __launch_bounds__(128)
void mha_gemm_wmma(const half_t* __restrict__ A,
                   const half_t* __restrict__ Bp,
                   half_t* __restrict__ C16, float* __restrict__ C32,
                   int N, int K) {
    __shared__ half_t Bs[2][2048];   // 2 x (4 subtiles x 512 halves) = 8KB

    int tid = threadIdx.x, lane = tid & 31, w = tid >> 5;
    int lo = lane & 15, hi = lane >> 4;
    int mt = blockIdx.x, ng = blockIdx.y;
    int NK = K >> 5;

    const half_t* arow  = A + (size_t)(mt * 64 + w * 16 + lo) * K;
    const half_t* bbase = Bp + (size_t)ng * NK * 2048;

    v8f acc[4];
#pragma unroll
    for (int j = 0; j < 4; ++j)
#pragma unroll
        for (int v = 0; v < 8; ++v) acc[j][v] = 0.0f;

    // prologue: stage chunk 0 (each thread copies 32B of the 4KB chunk)
    {
        const half_t* src = bbase + tid * 16;
        half_t* dst = &Bs[0][tid * 16];
        ASYNC_B128(src, dst);
        ASYNC_B128(src + 8, dst + 8);
    }

    for (int kt = 0; kt < NK; ++kt) {
        int cur = kt & 1;
        if (kt + 1 < NK) {
            const half_t* src = bbase + (size_t)(kt + 1) * 2048 + tid * 16;
            half_t* dst = &Bs[cur ^ 1][tid * 16];
            ASYNC_B128(src, dst);
            ASYNC_B128(src + 8, dst + 8);
            ASYNC_WAIT2();          // chunk kt fully landed (in-order)
        } else {
            ASYNC_WAIT0();
        }
        __syncthreads();

        // A fragment for this k-step
        int kb = (kt << 5) + (hi << 4);
        v8h c0 = *(const v8h*)(arow + kb);
        v8h c1 = *(const v8h*)(arow + kb + 8);
        v16h a;
#pragma unroll
        for (int i = 0; i < 8; ++i) { a[i] = c0[i]; a[i + 8] = c1[i]; }

        // preload all four B fragments (one clause of ds_loads, one wait)
        v16h b0 = *(const v16h*)(&Bs[cur][0 * 512 + lane * 16]);
        v16h b1 = *(const v16h*)(&Bs[cur][1 * 512 + lane * 16]);
        v16h b2 = *(const v16h*)(&Bs[cur][2 * 512 + lane * 16]);
        v16h b3 = *(const v16h*)(&Bs[cur][3 * 512 + lane * 16]);

        acc[0] = __builtin_amdgcn_wmma_f32_16x16x32_f16(
            false, a, false, b0, (short)0, acc[0], false, false);
        acc[1] = __builtin_amdgcn_wmma_f32_16x16x32_f16(
            false, a, false, b1, (short)0, acc[1], false, false);
        acc[2] = __builtin_amdgcn_wmma_f32_16x16x32_f16(
            false, a, false, b2, (short)0, acc[2], false, false);
        acc[3] = __builtin_amdgcn_wmma_f32_16x16x32_f16(
            false, a, false, b3, (short)0, acc[3], false, false);

        __syncthreads();   // all waves done reading before buffer reuse
    }

    int rbase = mt * 64 + w * 16 + hi * 8;
#pragma unroll
    for (int j = 0; j < 4; ++j) {
        int col = ng * 64 + j * 16 + lo;
        if (C32) {
#pragma unroll
            for (int v = 0; v < 8; ++v)
                C32[(size_t)(rbase + v) * N + col] = acc[j][v];
        } else {
#pragma unroll
            for (int v = 0; v < 8; ++v)
                C16[(size_t)(rbase + v) * N + col] = (half_t)acc[j][v];
        }
    }
}

// ---------------------------------------------------------------------------
// Flash attention: one wave handles a 32-row Q block (two 16-row tiles) for
// one (batch, head). K fragments and the LDS V tile are reused across both
// Q tiles; V is staged with async DMA overlapped with the QK^T WMMAs.
// grid = (SEQ/32, NHEADS, BATCH), block = 32
// ---------------------------------------------------------------------------
__global__ __launch_bounds__(32)
void mha_attention_wmma(const half_t* __restrict__ qkv,
                        half_t* __restrict__ out) {
    __shared__ half_t Vs[32 * 64];      // V tile (32 K-rows x 64 dh) = 4KB
    __shared__ half_t Ps[2][16 * 32];   // P tiles, one per Q subtile = 2KB

    int qt = blockIdx.x, h = blockIdx.y, b = blockIdx.z, lane = threadIdx.x;
    int lo = lane & 15, hi = lane >> 4;

    const half_t* base = qkv + (size_t)b * SEQ * TD;
    const half_t* qb  = base + h * DHEAD;
    const half_t* kbp = base + DMODEL + h * DHEAD;
    const half_t* vbp = base + 2 * DMODEL + h * DHEAD;

    // Q A-fragments: 2 subtiles x 2 dh-chunks
    v16h qa[2][2];
#pragma unroll
    for (int qi = 0; qi < 2; ++qi) {
        const half_t* qp = qb + (size_t)(qt * 32 + qi * 16 + lo) * TD + hi * 16;
        v8h a0 = *(const v8h*)(qp);
        v8h a1 = *(const v8h*)(qp + 8);
        v8h a2 = *(const v8h*)(qp + 32);
        v8h a3 = *(const v8h*)(qp + 40);
#pragma unroll
        for (int i = 0; i < 8; ++i) {
            qa[qi][0][i] = a0[i]; qa[qi][0][i + 8] = a1[i];
            qa[qi][1][i] = a2[i]; qa[qi][1][i + 8] = a3[i];
        }
    }

    float m[2][8], l[2][8];
    v8f o[2][4];
#pragma unroll
    for (int qi = 0; qi < 2; ++qi) {
#pragma unroll
        for (int v = 0; v < 8; ++v) { m[qi][v] = -INFINITY; l[qi][v] = 0.0f; }
#pragma unroll
        for (int g = 0; g < 4; ++g)
#pragma unroll
            for (int v = 0; v < 8; ++v) o[qi][g][v] = 0.0f;
    }

    for (int kt = 0; kt < SEQ / 32; ++kt) {
        int k0 = kt * 32;

        // --- stage V tile (lane -> one 64-half row) via async DMA ---
        {
            const half_t* vr = vbp + (size_t)(k0 + lane) * TD;
            half_t* vd = Vs + lane * 64;
#pragma unroll
            for (int j = 0; j < 8; ++j)
                ASYNC_B128(vr + j * 8, vd + j * 8);
        }

        // --- scores: K fragments shared across both Q subtiles ---
        v8f s[2][2];
#pragma unroll
        for (int t = 0; t < 2; ++t) {
            const half_t* kp = kbp + (size_t)(k0 + t * 16 + lo) * TD + hi * 16;
            v16h b0 = *(const v16h*)(kp);
            v16h b1 = *(const v16h*)(kp + 32);
#pragma unroll
            for (int qi = 0; qi < 2; ++qi) {
                v8f sc = {};
                sc = __builtin_amdgcn_wmma_f32_16x16x32_f16(
                    false, qa[qi][0], false, b0, (short)0, sc, false, false);
                sc = __builtin_amdgcn_wmma_f32_16x16x32_f16(
                    false, qa[qi][1], false, b1, (short)0, sc, false, false);
#pragma unroll
                for (int v = 0; v < 8; ++v) s[qi][t][v] = sc[v] * 0.125f;
            }
        }

        // --- online softmax per Q subtile ---
#pragma unroll
        for (int qi = 0; qi < 2; ++qi) {
            float al[8];
#pragma unroll
            for (int v = 0; v < 8; ++v) {
                float t = fmaxf(s[qi][0][v], s[qi][1][v]);
                t = fmaxf(t, __shfl_xor(t, 1, 16));
                t = fmaxf(t, __shfl_xor(t, 2, 16));
                t = fmaxf(t, __shfl_xor(t, 4, 16));
                t = fmaxf(t, __shfl_xor(t, 8, 16));
                float mnew = fmaxf(m[qi][v], t);
                float p0 = __expf(s[qi][0][v] - mnew);
                float p1 = __expf(s[qi][1][v] - mnew);
                float rs = p0 + p1;
                rs += __shfl_xor(rs, 1, 16);
                rs += __shfl_xor(rs, 2, 16);
                rs += __shfl_xor(rs, 4, 16);
                rs += __shfl_xor(rs, 8, 16);
                al[v] = __expf(m[qi][v] - mnew);
                l[qi][v] = l[qi][v] * al[v] + rs;
                m[qi][v] = mnew;
                Ps[qi][(hi * 8 + v) * 32 + lo]      = (half_t)p0;
                Ps[qi][(hi * 8 + v) * 32 + 16 + lo] = (half_t)p1;
            }
#pragma unroll
            for (int g = 0; g < 4; ++g)
#pragma unroll
                for (int v = 0; v < 8; ++v) o[qi][g][v] *= al[v];
        }

        ASYNC_WAIT0();      // V tile landed in LDS
        __syncthreads();    // Ps + Vs visible

        // --- P A-fragments ---
        v16h pa[2];
#pragma unroll
        for (int qi = 0; qi < 2; ++qi) {
            const half_t* pr = &Ps[qi][lo * 32 + hi * 16];
            v8h x0 = *(const v8h*)(pr);
            v8h x1 = *(const v8h*)(pr + 8);
#pragma unroll
            for (int i = 0; i < 8; ++i) { pa[qi][i] = x0[i]; pa[qi][i + 8] = x1[i]; }
        }

        // --- O += P * V; V fragments preloaded in pairs, reused across Q ---
#pragma unroll
        for (int gp = 0; gp < 2; ++gp) {
            v16h vva, vvb;
#pragma unroll
            for (int i = 0; i < 16; ++i) {
                vva[i] = Vs[(hi * 16 + i) * 64 + (gp * 2 + 0) * 16 + lo];
                vvb[i] = Vs[(hi * 16 + i) * 64 + (gp * 2 + 1) * 16 + lo];
            }
#pragma unroll
            for (int qi = 0; qi < 2; ++qi) {
                o[qi][gp * 2 + 0] = __builtin_amdgcn_wmma_f32_16x16x32_f16(
                    false, pa[qi], false, vva, (short)0, o[qi][gp * 2 + 0], false, false);
                o[qi][gp * 2 + 1] = __builtin_amdgcn_wmma_f32_16x16x32_f16(
                    false, pa[qi], false, vvb, (short)0, o[qi][gp * 2 + 1], false, false);
            }
        }
        __syncthreads();    // protect Vs/Ps before next iteration
    }

    // --- epilogue: normalize and store f16 [B,S,DMODEL] ---
#pragma unroll
    for (int qi = 0; qi < 2; ++qi)
#pragma unroll
        for (int g = 0; g < 4; ++g)
#pragma unroll
            for (int v = 0; v < 8; ++v) {
                size_t row = (size_t)b * SEQ + qt * 32 + qi * 16 + hi * 8 + v;
                out[row * DMODEL + h * DHEAD + g * 16 + lo] =
                    (half_t)(o[qi][g][v] / l[qi][v]);
            }
}

// ---------------------------------------------------------------------------
extern "C" void kernel_launch(void* const* d_in, const int* in_sizes, int n_in,
                              void* d_out, int out_size, void* d_ws, size_t ws_size,
                              hipStream_t stream) {
    (void)in_sizes; (void)n_in; (void)out_size; (void)ws_size;
    const float* x    = (const float*)d_in[0];   // [4,2048,1024]
    const float* Wqkv = (const float*)d_in[1];   // [1024,3072]
    const float* Wo   = (const float*)d_in[2];   // [1024,1024]
    float* out = (float*)d_out;                  // [4,2048,1024]

    const size_t MS = (size_t)BATCH * SEQ;       // 8192 rows
    half_t* xh     = (half_t*)d_ws;                      // 8192*1024
    half_t* wqkv_p = xh + MS * DMODEL;                   // 1024*3072
    half_t* wo_p   = wqkv_p + (size_t)DMODEL * TD;       // 1024*1024
    half_t* qkv_h  = wo_p + (size_t)DMODEL * DMODEL;     // 8192*3072
    half_t* attn_h = xh;   // reuse xh region after QKV GEMM consumes it

    // 1) x -> f16
    int nx = (int)(MS * DMODEL);
    mha_cvt_f32_to_f16<<<(nx + 255) / 256, 256, 0, stream>>>(x, xh, nx);

    // 2) repack weights into grouped B-fragment order
    mha_repack_b<<<dim3(DMODEL / 32, TD / 16), 32, 0, stream>>>(Wqkv, wqkv_p, DMODEL, TD);
    mha_repack_b<<<dim3(DMODEL / 32, DMODEL / 16), 32, 0, stream>>>(Wo, wo_p, DMODEL, DMODEL);

    // 3) qkv = x @ W_qkv  (f16 out)
    mha_gemm_wmma<<<dim3(MS / 64, TD / 64), 128, 0, stream>>>(
        xh, wqkv_p, qkv_h, nullptr, TD, DMODEL);

    // 4) flash attention (32 Q rows per wave)
    mha_attention_wmma<<<dim3(SEQ / 32, NHEADS, BATCH), 32, 0, stream>>>(qkv_h, attn_h);

    // 5) out = attn @ W_o  (f32 out)
    mha_gemm_wmma<<<dim3(MS / 64, DMODEL / 64), 128, 0, stream>>>(
        attn_h, wo_p, nullptr, out, DMODEL, DMODEL);
}
